// MolSSDDiffusion_23012434772662
// MI455X (gfx1250) — compile-verified, matrix operand order checked by reading, not run on hardware
//
#include <hip/hip_runtime.h>

#define NA 200000
#define NCC 512
#define NTYPES 5
#define KLAN 512
#define EPSF 1e-12f

typedef __attribute__((ext_vector_type(16))) __bf16 v16bf;
typedef __attribute__((ext_vector_type(8)))  float  v8f;

// Fragment-tile layout helpers: a K-tile is 16(m) x 32(kk) bf16 stored as
// 32 lanes x 16 elems (1KB), matching the documented 16-bit A/B WMMA VGPR map:
//   lane = m + 16*((kk>>3)&1),  elem = 8*(kk>>4) + (kk&7)
// Tile (nt, kt) of Q^T lives at ((nt*16 + kt) * 512) elems; same for S with jt.

// ---------------- init: zero scratch + diffusion schedule scalars ----------------
// scal: [0]=ab_t [1]=ab_tm1 [2]=sig2_t [3]=sig2_tm1 [4]=sigma_t [5]=a
//       [6]=norm2(v0) acc [7]=1/||v0|| [8]=alpha acc [9]=bn2 acc [10]=beta [11]=scale
__global__ void k_init(float* scal, float* zbase, int zcount, const int* t_ptr) {
    int tid = blockIdx.x * blockDim.x + threadIdx.x;
    for (int i = tid; i < zcount; i += gridDim.x * blockDim.x) zbase[i] = 0.f;
    if (blockIdx.x == 0 && threadIdx.x == 0) {
        float tf = (float)(*t_ptr);
        const float HPI = 1.5707963267948966f;
        float ab_t   = cosf(((tf        / 1000.f) + 0.008f) / 1.008f * HPI);
        float ab_tm1 = cosf((((tf - 1.f)/ 1000.f) + 0.008f) / 1.008f * HPI);
        float sig2_t   = 1.f - ab_t * ab_t;
        float sig2_tm1 = 1.f - ab_tm1 * ab_tm1;
        scal[0] = ab_t;  scal[1] = ab_tm1;
        scal[2] = sig2_t; scal[3] = sig2_tm1;
        scal[4] = sqrtf(sig2_t);
        scal[5] = ab_t / fmaxf(ab_tm1, EPSF);
        for (int j = 6; j < 16; j++) scal[j] = 0.f;
    }
}

// ---------------- pass over atoms: counts, cluster sums, type hist, ||v0||^2 ----
__global__ void __launch_bounds__(256) k_atoms1(
        const float* x0, const float* x0h, const float* v0,
        const int* types, const int* clus,
        float* counts, float* sum0, float* sum0h, float* hist, float* scal) {
    __shared__ float red[256];
    int i = blockIdx.x * 256 + threadIdx.x;
    float p = 0.f;
    if (i < NA) {
        int c = clus[i];
        atomicAdd(&counts[c], 1.f);
        atomicAdd(&hist[c * NTYPES + types[i]], 1.f);
        #pragma unroll
        for (int d = 0; d < 3; d++) {
            atomicAdd(&sum0 [c * 3 + d], x0 [i * 3 + d]);
            atomicAdd(&sum0h[c * 3 + d], x0h[i * 3 + d]);
        }
        float vv = v0[i];
        p = vv * vv;
    }
    red[threadIdx.x] = p; __syncthreads();
    for (int s = 128; s > 0; s >>= 1) {
        if (threadIdx.x < s) red[threadIdx.x] += red[threadIdx.x + s];
        __syncthreads();
    }
    if (threadIdx.x == 0) atomicAdd(&scal[6], red[0]);
}

// ---------------- per-cluster finalize: x_t, residual coeff, argmax types -------
__global__ void __launch_bounds__(512) k_fin_clusters(
        const float* eps_in, float* counts, const float* sum0, const float* sum0h,
        const float* hist, float* scal, float* mucoef,
        float* xt_out, float* type_out) {
    int c = threadIdx.x;
    if (c == 0) scal[7] = 1.f / sqrtf(fmaxf(scal[6], 1e-30f));
    float ab_t = scal[0], ab_tm1 = scal[1], sig2_t = scal[2], sig2_tm1 = scal[3];
    float sigma_t = scal[4], a = scal[5];
    float cnt = fmaxf(counts[c], 1.f);
    counts[c] = cnt;
    float coef = (sig2_tm1 / fmaxf(sig2_t, EPSF)) * a / cnt;
    #pragma unroll
    for (int d = 0; d < 3; d++) {
        float xm = sum0[c * 3 + d] / cnt;
        float xt = ab_t * xm + sigma_t * eps_in[c * 3 + d];
        xt_out[c * 3 + d] = xt;
        float cm = ab_tm1 * sum0h[c * 3 + d] / cnt;   // C_apply(ab_tm1 * x0_hat)
        mucoef[c * 3 + d] = coef * (xt - a * cm);
    }
    float best = hist[c * NTYPES]; int barg = 0;
    for (int j = 1; j < NTYPES; j++) {
        float h = hist[c * NTYPES + j];
        if (h > best) { best = h; barg = j; }
    }
    type_out[c] = (float)barg;
}

// ---------------- mu per atom + Lanczos v1 / v_prev init ------------------------
__global__ void __launch_bounds__(256) k_mu_vinit(
        const float* x0h, const int* clus, const float* mucoef, const float* scal,
        const float* v0, float* mu_out, float* v, float* vprev) {
    int i = blockIdx.x * 256 + threadIdx.x;
    if (i >= NA) return;
    int c = clus[i];
    float ab_tm1 = scal[1];
    #pragma unroll
    for (int d = 0; d < 3; d++)
        mu_out[i * 3 + d] = ab_tm1 * x0h[i * 3 + d] + mucoef[c * 3 + d];
    v[i] = v0[i] * scal[7];
    vprev[i] = 0.f;
}

// ---------------- Lanczos: segment sum of v over clusters -----------------------
__global__ void __launch_bounds__(256) k_segsum(const float* v, const int* clus, float* sseg) {
    __shared__ float ls[NCC];
    int t = threadIdx.x;
    for (int j = t; j < NCC; j += 256) ls[j] = 0.f;
    __syncthreads();
    int i = blockIdx.x * 256 + t;
    if (i < NA) atomicAdd(&ls[clus[i]], v[i]);
    __syncthreads();
    for (int j = t; j < NCC; j += 256) {
        float val = ls[j];
        if (val != 0.f) atomicAdd(&sseg[j], val);
    }
}

// ---------------- Lanczos: w = A v - beta v_prev ; alpha = <v,w> ----------------
__global__ void __launch_bounds__(256) k_wstep1(
        const float* v, const float* vprev, const int* clus,
        const float* sseg, const float* counts, float* scal, float* w) {
    __shared__ float red[256];
    int i = blockIdx.x * 256 + threadIdx.x;
    float p = 0.f;
    if (i < NA) {
        int c = clus[i];
        float a = scal[5], beta = scal[10];
        float cnt = counts[c];
        float wi = (a * a) * sseg[c] / (cnt * cnt) - beta * vprev[i];
        w[i] = wi;
        p = v[i] * wi;
    }
    red[threadIdx.x] = p; __syncthreads();
    for (int s = 128; s > 0; s >>= 1) {
        if (threadIdx.x < s) red[threadIdx.x] += red[threadIdx.x + s];
        __syncthreads();
    }
    if (threadIdx.x == 0) atomicAdd(&scal[8], red[0]);
}

// ------- Lanczos: w -= alpha v ; bn2 = ||w||^2 ; scatter Q[k] into frag tiles ---
__global__ void __launch_bounds__(256) k_wstep2(
        const float* v, float* w, float* scal, __bf16* Qf, int krow) {
    __shared__ float red[256];
    int i = blockIdx.x * 256 + threadIdx.x;
    float p = 0.f;
    // tile coords for this Lanczos row are launch-uniform
    int kt = krow >> 5, kk = krow & 31;
    int bh = (kk >> 3) & 1, hh = kk >> 4, rr = kk & 7;
    if (i < NA) {
        float al = scal[8];
        float wi = w[i] - al * v[i];
        w[i] = wi;
        p = wi * wi;
        int ntile = i >> 4, mm = i & 15;
        size_t off = ((size_t)ntile * 16 + kt) * 512
                   + (size_t)(mm + 16 * bh) * 16 + 8 * hh + rr;
        Qf[off] = (__bf16)v[i];
    }
    red[threadIdx.x] = p; __syncthreads();
    for (int s = 128; s > 0; s >>= 1) {
        if (threadIdx.x < s) red[threadIdx.x] += red[threadIdx.x + s];
        __syncthreads();
    }
    if (threadIdx.x == 0) atomicAdd(&scal[9], red[0]);
}

// ---------------- Lanczos: finalize iteration scalars, re-zero accumulators -----
__global__ void __launch_bounds__(512) k_lfin(float* scal, float* alphas, float* betas,
                                              float* sseg, int krow) {
    int t = threadIdx.x;
    if (t == 0) {
        float al = scal[8], bn2 = scal[9];
        float bn = sqrtf(bn2);
        bool good = bn > 1e-8f;
        alphas[krow] = al;
        float bnm = good ? bn : 0.f;
        betas[krow] = bnm;
        scal[10] = bnm;
        scal[11] = good ? 1.f / fmaxf(bn, EPSF) : 0.f;
        scal[8] = 0.f; scal[9] = 0.f;
    }
    sseg[t] = 0.f;   // ready for next iteration's segment sum
}

__global__ void __launch_bounds__(256) k_vnext(float* w, const float* scal) {
    int i = blockIdx.x * 256 + threadIdx.x;
    if (i < NA) w[i] *= scal[11];
}

// ---------------- single-WGP parallel TQL2 eigensolver (512x512 tridiag) --------
__global__ void __launch_bounds__(512) k_eigh(
        const float* alphas, const float* betas, const float* scal,
        float* Z, __bf16* Sf, float* eig_out) {
    __shared__ float d[KLAN], e[KLAN], cs[KLAN], sn[KLAN];
    __shared__ int sh_m, sh_ilow, perm[KLAN];
    int t = threadIdx.x;
    d[t] = alphas[t];
    e[t] = (t < KLAN - 1) ? betas[t] : 0.f;
    long rb = (long)t * KLAN;
    for (int j = 0; j < KLAN; j++) Z[rb + j] = (t == j) ? 1.f : 0.f;
    __syncthreads();

    for (int l = 0; l < KLAN; l++) {
        for (int it = 0; it < 64; it++) {
            if (t == 0) {
                int m;
                for (m = l; m < KLAN - 1; m++) {
                    float dd = fabsf(d[m]) + fabsf(d[m + 1]);
                    if (fabsf(e[m]) <= 1.19209290e-7f * dd) break;
                }
                sh_m = m; sh_ilow = l;
                if (m != l) {
                    float g = (d[l + 1] - d[l]) / (2.f * e[l]);
                    float r = sqrtf(g * g + 1.f);
                    g = d[m] - d[l] + e[l] / (g + copysignf(r, g));
                    float ss = 1.f, c = 1.f, p = 0.f;
                    int i; bool broke = false;
                    for (i = m - 1; i >= l; i--) {
                        float f = ss * e[i], b = c * e[i];
                        r = sqrtf(f * f + g * g);
                        e[i + 1] = r;
                        if (r == 0.f) { d[i + 1] -= p; e[m] = 0.f; broke = true; break; }
                        ss = f / r; c = g / r;
                        g = d[i + 1] - p;
                        r = (d[i] - g) * ss + 2.f * c * b;
                        p = ss * r;
                        d[i + 1] = g + p;
                        g = c * r - b;
                        cs[i] = c; sn[i] = ss;
                    }
                    if (!broke) { d[l] -= p; e[l] = g; e[m] = 0.f; sh_ilow = l; }
                    else sh_ilow = i + 1;
                }
            }
            __syncthreads();
            int m = sh_m;
            if (m == l) break;
            int ilow = sh_ilow;
            // each thread applies rotation chain to its own row of Z
            for (int i = m - 1; i >= ilow; i--) {
                float c = cs[i], ss = sn[i];
                float f  = Z[rb + i + 1];
                float zi = Z[rb + i];
                Z[rb + i + 1] = ss * zi + c * f;
                Z[rb + i]     = c * zi - ss * f;
            }
            __syncthreads();
        }
    }
    __syncthreads();
    // rank sort (ascending, ties by index) to match eigh ordering
    float dj = d[t];
    int rank = 0;
    for (int i = 0; i < KLAN; i++) {
        float di = d[i];
        if (di < dj || (di == dj && i < t)) rank++;
    }
    perm[rank] = t;
    float sig2_t = scal[2], sig2_tm1 = scal[3];
    eig_out[rank] = fmaxf(sig2_tm1 - (sig2_tm1 * sig2_tm1 / fmaxf(sig2_t, EPSF)) * dj, 0.f);
    __syncthreads();
    // write permuted eigenvector matrix straight into B-fragment tile layout
    // element (k=t, j): tile (jt, kt); lane = (j&15)+16*((kk>>3)&1); e = 8*(kk>>4)+(kk&7)
    int kt = t >> 5, kk = t & 31;
    int bh = (kk >> 3) & 1, hh = kk >> 4, rr2 = kk & 7;
    size_t laneoff = (size_t)(8 * hh + rr2) + (size_t)16 * 16 * bh; // + (jm)*16 added below
    for (int j = 0; j < KLAN; j++) {
        float val = Z[rb + perm[j]];
        int jt = j >> 4, jm = j & 15;
        size_t off = ((size_t)jt * 16 + kt) * 512 + (size_t)jm * 16 + laneoff;
        Sf[off] = (__bf16)val;
    }
}

// ---------- ritz = Q^T * S, pure global->WMMA (fragments pre-laid-out) ----------
// 256 threads = 8 waves; block owns one 16-row n-tile, each wave owns 4 j-tiles
// (4x A-register reuse). Inner loop: 5 x (2x global_load_b128) + 4 x wmma.
__global__ void __launch_bounds__(256) k_gemm(const __bf16* __restrict__ Qf,
                                              const __bf16* __restrict__ Sf,
                                              float* __restrict__ out) {
    int tid = threadIdx.x;
    int wave = tid >> 5, lane = tid & 31;
    int ntile = blockIdx.x;
    int jt0 = wave << 2;                 // 4 j-tiles per wave, 8 waves = 32 tiles
    const __bf16* aB  = Qf + (size_t)ntile * 16 * 512 + (size_t)lane * 16;
    const __bf16* bB0 = Sf + (size_t)(jt0 + 0) * 16 * 512 + (size_t)lane * 16;
    const __bf16* bB1 = bB0 + 16 * 512;
    const __bf16* bB2 = bB1 + 16 * 512;
    const __bf16* bB3 = bB2 + 16 * 512;
    v8f acc0 = {}, acc1 = {}, acc2 = {}, acc3 = {};
    #pragma unroll 4
    for (int kt = 0; kt < 16; kt++) {
        size_t o = (size_t)kt * 512;
        v16bf a  = *(const v16bf*)(aB  + o);
        v16bf b0 = *(const v16bf*)(bB0 + o);
        v16bf b1 = *(const v16bf*)(bB1 + o);
        v16bf b2 = *(const v16bf*)(bB2 + o);
        v16bf b3 = *(const v16bf*)(bB3 + o);
        acc0 = __builtin_amdgcn_wmma_f32_16x16x32_bf16(false, a, false, b0, (short)0, acc0, false, false);
        acc1 = __builtin_amdgcn_wmma_f32_16x16x32_bf16(false, a, false, b1, (short)0, acc1, false, false);
        acc2 = __builtin_amdgcn_wmma_f32_16x16x32_bf16(false, a, false, b2, (short)0, acc2, false, false);
        acc3 = __builtin_amdgcn_wmma_f32_16x16x32_bf16(false, a, false, b3, (short)0, acc3, false, false);
    }
    // C/D layout: col = lane&15, row = r + 8*(lane>=16)
    int row0 = ntile * 16 + ((lane >> 4) << 3);
    int colb = (jt0 << 4) + (lane & 15);
    #pragma unroll
    for (int r = 0; r < 8; r++) {
        float* orow = out + (size_t)(row0 + r) * KLAN + colb;
        orow[0]  = acc0[r];
        orow[16] = acc1[r];
        orow[32] = acc2[r];
        orow[48] = acc3[r];
    }
}

// =================================================================================
extern "C" void kernel_launch(void* const* d_in, const int* in_sizes, int n_in,
                              void* d_out, int out_size, void* d_ws, size_t ws_size,
                              hipStream_t stream) {
    const float* x0    = (const float*)d_in[0];
    const float* x0h   = (const float*)d_in[1];
    const float* eps   = (const float*)d_in[2];
    const float* v0    = (const float*)d_in[3];
    const int*   types = (const int*)d_in[4];
    const int*   clus  = (const int*)d_in[5];
    const int*   t_ptr = (const int*)d_in[6];

    float* out = (float*)d_out;
    float* xt_out   = out;                                    // 512*3
    float* mu_out   = out + 1536;                             // 200000*3
    float* eig_out  = out + 601536;                           // 512
    float* ritz_out = out + 602048;                           // 200000*512
    float* type_out = out + 602048 + (long)NA * KLAN;         // 512

    char* cur = (char*)d_ws;
    auto alloc = [&](size_t bytes) -> void* {
        void* p = (void*)cur;
        cur += (bytes + 255) & ~(size_t)255;
        return p;
    };
    float* scal   = (float*)alloc(64 * 4);
    float* counts = (float*)alloc(NCC * 4);
    float* sum0   = (float*)alloc(NCC * 3 * 4);
    float* sum0h  = (float*)alloc(NCC * 3 * 4);
    float* hist   = (float*)alloc(NCC * NTYPES * 4);
    float* sseg   = (float*)alloc(NCC * 4);
    float* alphas = (float*)alloc(KLAN * 4);
    float* betas  = (float*)alloc(KLAN * 4);
    float* mucoef = (float*)alloc(NCC * 3 * 4);
    float* vA     = (float*)alloc((size_t)NA * 4);
    float* vB     = (float*)alloc((size_t)NA * 4);
    float* vC     = (float*)alloc((size_t)NA * 4);
    float* Z      = (float*)alloc((size_t)KLAN * KLAN * 4);
    __bf16* Sf    = (__bf16*)alloc((size_t)KLAN * KLAN * 2);
    __bf16* Qf    = (__bf16*)alloc((size_t)KLAN * NA * 2);

    int zcount = (int)(((char*)mucoef + NCC * 3 * 4) - (char*)counts) / 4;
    const int NB = (NA + 255) / 256;  // 782

    k_init<<<32, 256, 0, stream>>>(scal, counts, zcount, t_ptr);
    k_atoms1<<<NB, 256, 0, stream>>>(x0, x0h, v0, types, clus,
                                     counts, sum0, sum0h, hist, scal);
    k_fin_clusters<<<1, 512, 0, stream>>>(eps, counts, sum0, sum0h, hist,
                                          scal, mucoef, xt_out, type_out);
    k_mu_vinit<<<NB, 256, 0, stream>>>(x0h, clus, mucoef, scal, v0, mu_out, vB, vA);

    float* vp = vA; float* v = vB; float* wv = vC;
    for (int k = 0; k < KLAN; k++) {
        k_segsum<<<NB, 256, 0, stream>>>(v, clus, sseg);
        k_wstep1<<<NB, 256, 0, stream>>>(v, vp, clus, sseg, counts, scal, wv);
        k_wstep2<<<NB, 256, 0, stream>>>(v, wv, scal, Qf, k);
        k_lfin<<<1, 512, 0, stream>>>(scal, alphas, betas, sseg, k);
        k_vnext<<<NB, 256, 0, stream>>>(wv, scal);
        float* tmp = vp; vp = v; v = wv; wv = tmp;
    }

    k_eigh<<<1, 512, 0, stream>>>(alphas, betas, scal, Z, Sf, eig_out);

    k_gemm<<<NA / 16, 256, 0, stream>>>(Qf, Sf, ritz_out);   // 12500 blocks
}
